// BiasedMultiheadAttention_20134806684330
// MI455X (gfx1250) — matrix operand order repaired and checked by reading, not compiled, not run on hardware
//
#include <hip/hip_runtime.h>

constexpr int cB = 2, cT = 2048, cD = 1024, cH = 16, cDM = 256, cDK = 64;

typedef _Float16 v8h  __attribute__((ext_vector_type(8)));
typedef _Float16 v16h __attribute__((ext_vector_type(16)));
typedef float    v8f  __attribute__((ext_vector_type(8)));
typedef int      v4i  __attribute__((ext_vector_type(4)));

// ---- CDNA5 async global->LDS copy (ASYNCcnt), guarded so we degrade to the
// ---- synchronous path if this toolchain doesn't declare the builtins.
#if defined(__HIP_DEVICE_COMPILE__) && defined(__gfx1250__) && \
    __has_builtin(__builtin_amdgcn_global_load_async_to_lds_b128)
#define HAVE_ASYNC_LDS 1
typedef __attribute__((address_space(1))) v4i v4i_g;   // global int4
typedef __attribute__((address_space(3))) v4i v4i_l;   // LDS int4
__device__ __forceinline__ void async_cp16(const _Float16* g, _Float16* l) {
  __builtin_amdgcn_global_load_async_to_lds_b128(
      (v4i_g*)(void*)g, (v4i_l*)(void*)l, 0, 0);
}
__device__ __forceinline__ void async_cp16f(const float* g, float* l) {
  __builtin_amdgcn_global_load_async_to_lds_b128(
      (v4i_g*)(void*)g, (v4i_l*)(void*)l, 0, 0);
}
__device__ __forceinline__ void wait_async0() {
#if __has_builtin(__builtin_amdgcn_s_wait_asynccnt)
  __builtin_amdgcn_s_wait_asynccnt(0);
#else
  asm volatile("s_wait_asynccnt 0x0" ::: "memory");
#endif
}
#else
#define HAVE_ASYNC_LDS 0
#endif

// Build a 16-half fragment from two 16-byte chunks (A-frag and B-frag: one
// contiguous 16-half run K=16*hi..16*hi+15 split into two aligned b128 loads).
__device__ __forceinline__ v16h frag_ld(const _Float16* p0, const _Float16* p1) {
  v8h a = *(const v8h*)p0;
  v8h b = *(const v8h*)p1;
  return __builtin_shufflevector(a, b, 0,1,2,3,4,5,6,7,8,9,10,11,12,13,14,15);
}

__device__ __forceinline__ v8f wmma32(v16h a, v16h b, v8f c) {
  return __builtin_amdgcn_wmma_f32_16x16x32_f16(false, a, false, b, (short)0, c,
                                                false, false);
}

__global__ void cvt_f32_f16(const float* __restrict__ s, _Float16* __restrict__ d, int n) {
  int i = blockIdx.x * 256 + threadIdx.x;
  if (i < n) d[i] = (_Float16)s[i];
}

// h_cat[b,t,:] = concat(Hx[b,:,t], Hf[b,t,:]) in f16.
__global__ void pack_hcat(const float* __restrict__ Hx, const float* __restrict__ Hf,
                          _Float16* __restrict__ Hcat) {
  size_t i = (size_t)blockIdx.x * 256 + threadIdx.x;
  if (i >= (size_t)cB * cT * cD) return;
  int dd = (int)(i % cD);
  size_t bt = i / cD;
  int t = (int)(bt % cT);
  int b = (int)(bt / cT);
  float x = Hx[((size_t)b * cD + dd) * cT + t];
  Hcat[bt * (2 * cD) + dd]      = (_Float16)x;
  Hcat[bt * (2 * cD) + cD + dd] = (_Float16)Hf[i];
}

// One 64-wide K step: 2 x (A-frag x 4 B-frags) = 8 WMMAs per wave.
__device__ __forceinline__ void mm_step(_Float16 (*Bs)[72],
                                        const _Float16* arow, int kk,
                                        int nloc, int n, int hi,
                                        v8f& a0, v8f& a1, v8f& a2, v8f& a3) {
#pragma unroll
  for (int c = 0; c < 2; ++c) {
    const int cb = 32 * c;
    v16h a = frag_ld(arow + kk + cb + 16 * hi, arow + kk + cb + 16 * hi + 8);
    v16h b0 = frag_ld(&Bs[nloc +      n][cb + 16 * hi], &Bs[nloc +      n][cb + 16 * hi + 8]);
    a0 = wmma32(a, b0, a0);
    v16h b1 = frag_ld(&Bs[nloc + 16 + n][cb + 16 * hi], &Bs[nloc + 16 + n][cb + 16 * hi + 8]);
    a1 = wmma32(a, b1, a1);
    v16h b2 = frag_ld(&Bs[nloc + 32 + n][cb + 16 * hi], &Bs[nloc + 32 + n][cb + 16 * hi + 8]);
    a2 = wmma32(a, b2, a2);
    v16h b3 = frag_ld(&Bs[nloc + 48 + n][cb + 16 * hi], &Bs[nloc + 48 + n][cb + 16 * hi + 8]);
    a3 = wmma32(a, b3, a3);
  }
}

// C(MxN) = A(MxK) @ W(NxK)^T + bias[N].  Block tile 64x128 (8 waves, 16x64
// each); B tile staged in LDS once per block, async double-buffered.
template <bool F16OUT>
__global__ void __launch_bounds__(256)
gemm_xwT(const _Float16* __restrict__ A, int lda,
         const _Float16* __restrict__ W,
         const float* __restrict__ bias,
         void* __restrict__ Cout, int ldc, int K) {
  __shared__ _Float16 Bs[2][128][72];

  const int tid  = threadIdx.x;
  const int wave = tid >> 5, lane = tid & 31;
  const int n = lane & 15, hi = lane >> 4;
  const int m0 = blockIdx.x * 64 + (wave >> 1) * 16;
  const int nloc = (wave & 1) * 64;
  const int n0 = blockIdx.y * 128 + nloc;

  const v8f Z = {0.f,0.f,0.f,0.f,0.f,0.f,0.f,0.f};
  v8f acc0 = Z, acc1 = Z, acc2 = Z, acc3 = Z;

  const _Float16* arow = A + (size_t)(m0 + n) * lda;
  const int trow = tid >> 1;                // 0..127: B tile row
  const int tcb  = (tid & 1) * 32;          // half-row base (in halfs)
  const _Float16* wrow = W + (size_t)(blockIdx.y * 128 + trow) * K + tcb;

#if HAVE_ASYNC_LDS
#pragma unroll
  for (int j = 0; j < 4; ++j)
    async_cp16(wrow + j * 8, &Bs[0][trow][tcb + j * 8]);
  for (int kk = 0, s = 0; kk < K; kk += 64, ++s) {
    wait_async0();
    __syncthreads();
    if (kk + 64 < K) {
#pragma unroll
      for (int j = 0; j < 4; ++j)
        async_cp16(wrow + kk + 64 + j * 8, &Bs[(s + 1) & 1][trow][tcb + j * 8]);
    }
    __builtin_prefetch(arow + kk + 256, 0, 1);
    mm_step(Bs[s & 1], arow, kk, nloc, n, hi, acc0, acc1, acc2, acc3);
  }
#else
  for (int kk = 0; kk < K; kk += 64) {
    __syncthreads();
#pragma unroll
    for (int j = 0; j < 4; ++j)
      *(v8h*)&Bs[0][trow][tcb + j * 8] = *(const v8h*)(wrow + kk + j * 8);
    __syncthreads();
    __builtin_prefetch(arow + kk + 256, 0, 1);
    mm_step(Bs[0], arow, kk, nloc, n, hi, acc0, acc1, acc2, acc3);
  }
#endif

  const float bv0 = bias[n0 + n];
  const float bv1 = bias[n0 + 16 + n];
  const float bv2 = bias[n0 + 32 + n];
  const float bv3 = bias[n0 + 48 + n];

#pragma unroll
  for (int r = 0; r < 8; ++r) {
    const size_t row = (size_t)(m0 + r + 8 * hi);
    if constexpr (F16OUT) {
      _Float16* C = (_Float16*)Cout;
      C[row * ldc + n0 +      n] = (_Float16)(acc0[r] + bv0);
      C[row * ldc + n0 + 16 + n] = (_Float16)(acc1[r] + bv1);
      C[row * ldc + n0 + 32 + n] = (_Float16)(acc2[r] + bv2);
      C[row * ldc + n0 + 48 + n] = (_Float16)(acc3[r] + bv3);
    } else {
      float* C = (float*)Cout;
      C[row * ldc + n0 +      n] = acc0[r] + bv0;
      C[row * ldc + n0 + 16 + n] = acc1[r] + bv1;
      C[row * ldc + n0 + 32 + n] = acc2[r] + bv2;
      C[row * ldc + n0 + 48 + n] = acc3[r] + bv3;
    }
  }
}

// Flash attention: block = 8 waves x 16 q-rows = 128 queries of one (b,h).
// K tiles + bias tiles async double-buffered in LDS; V staged once and
// transposed between the two barriers so a single raw buffer suffices.
__global__ void __launch_bounds__(256)
attn_fwd(const _Float16* __restrict__ Qh, const _Float16* __restrict__ Kh,
         const _Float16* __restrict__ Vh, const float* __restrict__ BiasM,
         _Float16* __restrict__ Oh) {
  __shared__ _Float16 Ks[2][32][72];   // keys x head-dim (ping-pong)
  __shared__ _Float16 Vr[32][72];      // raw V tile (single)
  __shared__ _Float16 VT[64][40];      // head-dim x keys (transposed)
  __shared__ _Float16 Ps[8][16][40];   // per-wave P layout shuffle
  __shared__ float    Bx[2][128][32];  // bias tile (ping-pong)

  const int blk = blockIdx.x;
  const int qt = blk & 15;             // T/128 = 16 q-tiles
  const int h  = (blk >> 4) & 15;
  const int b  = blk >> 8;
  const int tid = threadIdx.x;
  const int wave = tid >> 5, lane = tid & 31;
  const int n = lane & 15, hi = lane >> 4;
  const int q0 = qt * 128 + wave * 16;
  const size_t rowbase = (size_t)b * cT;
  const int colH = h * cDK;
  const int trow = tid >> 3, tcid = (tid & 7) * 8;   // K/V staging chunk
  const int brow = tid >> 1, bcb = (tid & 1) * 16;   // bias staging chunk
  const int qb = qt * 128;

  const _Float16* qrow = Qh + (rowbase + q0 + n) * cD + colH;
  v16h Aq0 = frag_ld(qrow + 16 * hi,      qrow + 16 * hi + 8);
  v16h Aq1 = frag_ld(qrow + 32 + 16 * hi, qrow + 32 + 16 * hi + 8);

  const v8f Z = {0.f,0.f,0.f,0.f,0.f,0.f,0.f,0.f};
  v8f O0 = Z, O1 = Z, O2 = Z, O3 = Z;
  float mrow[8], lrow[8];
#pragma unroll
  for (int r = 0; r < 8; ++r) { mrow[r] = -1e30f; lrow[r] = 0.f; }
  const float scale = 0.125f;          // 1/sqrt(64)

#if HAVE_ASYNC_LDS
  {
    const size_t g0 = (rowbase + trow) * cD + colH + tcid;
    async_cp16(Kh + g0, &Ks[0][trow][tcid]);
    async_cp16(Vh + g0, &Vr[trow][tcid]);
    const float* bsrc = BiasM + (rowbase + qb + brow) * (size_t)cT + bcb;
#pragma unroll
    for (int j = 0; j < 4; ++j)
      async_cp16f(bsrc + j * 4, &Bx[0][brow][bcb + j * 4]);
  }
#endif

  for (int kt = 0, s = 0; kt < cT; kt += 32, ++s) {
#if HAVE_ASYNC_LDS
    const int c = s & 1;
    wait_async0();
    __syncthreads();
    { // V^T from LDS raw tile (frees Vr for the next async fill)
      v8h vv = *(const v8h*)&Vr[trow][tcid];
#pragma unroll
      for (int j = 0; j < 8; ++j) VT[tcid + j][trow] = vv[j];
    }
    __syncthreads();
    if (kt + 32 < cT) {
      const size_t gn = (rowbase + kt + 32 + trow) * cD + colH + tcid;
      async_cp16(Kh + gn, &Ks[c ^ 1][trow][tcid]);
      async_cp16(Vh + gn, &Vr[trow][tcid]);
      const float* bsrc = BiasM + (rowbase + qb + brow) * (size_t)cT + kt + 32 + bcb;
#pragma unroll
      for (int j = 0; j < 4; ++j)
        async_cp16f(bsrc + j * 4, &Bx[c ^ 1][brow][bcb + j * 4]);
    }
#else
    const int c = 0;
    __syncthreads();
    {
      const size_t g = (rowbase + kt + trow) * cD + colH + tcid;
      *(v8h*)&Ks[0][trow][tcid] = *(const v8h*)(Kh + g);
      v8h vv = *(const v8h*)(Vh + g);
#pragma unroll
      for (int j = 0; j < 8; ++j) VT[tcid + j][trow] = vv[j];
    }
    __syncthreads();
#endif
    const _Float16 (*Kc)[72] = Ks[c];

    // S = Q K^T for 16 queries x 32 keys (two 16x16 accumulators)
    v8f S0 = Z, S1 = Z;
    {
      v16h bk;
      bk = frag_ld(&Kc[n][16 * hi],           &Kc[n][16 * hi + 8]);            S0 = wmma32(Aq0, bk, S0);
      bk = frag_ld(&Kc[n][32 + 16 * hi],      &Kc[n][32 + 16 * hi + 8]);       S0 = wmma32(Aq1, bk, S0);
      bk = frag_ld(&Kc[16 + n][16 * hi],      &Kc[16 + n][16 * hi + 8]);       S1 = wmma32(Aq0, bk, S1);
      bk = frag_ld(&Kc[16 + n][32 + 16 * hi], &Kc[16 + n][32 + 16 * hi + 8]);  S1 = wmma32(Aq1, bk, S1);
    }

    // scale + bias + online softmax (row stats replicated across 16-lane halves)
#pragma unroll
    for (int r = 0; r < 8; ++r) {
#if HAVE_ASYNC_LDS
      const float* bxr = Bx[c][wave * 16 + r + 8 * hi];
      float s0 = S0[r] * scale + bxr[n];
      float s1 = S1[r] * scale + bxr[16 + n];
#else
      const size_t br = (rowbase + q0 + r + 8 * hi) * (size_t)cT + kt;
      float s0 = S0[r] * scale + BiasM[br + n];
      float s1 = S1[r] * scale + BiasM[br + 16 + n];
#endif
      float rm = fmaxf(s0, s1);
#pragma unroll
      for (int msk = 8; msk > 0; msk >>= 1) rm = fmaxf(rm, __shfl_xor(rm, msk, 16));
      const float mnew = fmaxf(mrow[r], rm);
      const float p0 = __expf(s0 - mnew);
      const float p1 = __expf(s1 - mnew);
      float rs = p0 + p1;
#pragma unroll
      for (int msk = 8; msk > 0; msk >>= 1) rs += __shfl_xor(rs, msk, 16);
      const float corr = __expf(mrow[r] - mnew);
      lrow[r] = lrow[r] * corr + rs;
      mrow[r] = mnew;
      O0[r] *= corr; O1[r] *= corr; O2[r] *= corr; O3[r] *= corr;
      Ps[wave][r + 8 * hi][n]      = (_Float16)p0;   // C-layout -> LDS
      Ps[wave][r + 8 * hi][16 + n] = (_Float16)p1;
    }

    // O += P V  (A-frag of P via per-wave LDS; intra-wave DS ops are in-order)
    v16h Ap = frag_ld(&Ps[wave][n][16 * hi], &Ps[wave][n][16 * hi + 8]);
    v16h bvf;
    bvf = frag_ld(&VT[n][16 * hi],      &VT[n][16 * hi + 8]);       O0 = wmma32(Ap, bvf, O0);
    bvf = frag_ld(&VT[16 + n][16 * hi], &VT[16 + n][16 * hi + 8]);  O1 = wmma32(Ap, bvf, O1);
    bvf = frag_ld(&VT[32 + n][16 * hi], &VT[32 + n][16 * hi + 8]);  O2 = wmma32(Ap, bvf, O2);
    bvf = frag_ld(&VT[48 + n][16 * hi], &VT[48 + n][16 * hi + 8]);  O3 = wmma32(Ap, bvf, O3);
  }

#pragma unroll
  for (int r = 0; r < 8; ++r) {
    const float inv = 1.0f / lrow[r];
    _Float16* orow = Oh + (rowbase + q0 + r + 8 * hi) * cD + colH;
    orow[n]      = (_Float16)(O0[r] * inv);
    orow[16 + n] = (_Float16)(O1[r] * inv);
    orow[32 + n] = (_Float16)(O2[r] * inv);
    orow[48 + n] = (_Float16)(O3[r] * inv);
  }
}

extern "C" void kernel_launch(void* const* d_in, const int* in_sizes, int n_in,
                              void* d_out, int out_size, void* d_ws, size_t ws_size,
                              hipStream_t stream) {
  (void)in_sizes; (void)n_in; (void)out_size; (void)ws_size;
  const float* Hx = (const float*)d_in[0];
  const float* Hf = (const float*)d_in[1];
  const float* Gm = (const float*)d_in[2];
  const float* Wg = (const float*)d_in[3];
  const float* bg = (const float*)d_in[4];
  const float* Wq = (const float*)d_in[5];
  const float* bq = (const float*)d_in[6];
  const float* Wk = (const float*)d_in[7];
  const float* bk = (const float*)d_in[8];
  const float* Wv = (const float*)d_in[9];
  const float* bv = (const float*)d_in[10];
  const float* Wo = (const float*)d_in[11];
  const float* bo = (const float*)d_in[12];

  char* ws = (char*)d_ws;
  size_t off = 0;
  auto take = [&](size_t bytes) -> void* {
    void* p = ws + off;
    off = (off + bytes + 255) & ~(size_t)255;
    return p;
  };
  _Float16* WqH  = (_Float16*)take(sizeof(_Float16) * (size_t)cD * 2 * cD);
  _Float16* WkH  = (_Float16*)take(sizeof(_Float16) * (size_t)cD * 2 * cD);
  _Float16* WvH  = (_Float16*)take(sizeof(_Float16) * (size_t)cD * cD);
  _Float16* WoH  = (_Float16*)take(sizeof(_Float16) * (size_t)cD * cD);
  _Float16* WgH  = (_Float16*)take(sizeof(_Float16) * (size_t)cT * cDM);
  _Float16* GmH  = (_Float16*)take(sizeof(_Float16) * (size_t)cB * cT * cDM);
  _Float16* Hcat = (_Float16*)take(sizeof(_Float16) * (size_t)cB * cT * 2 * cD);
  _Float16* QH   = (_Float16*)take(sizeof(_Float16) * (size_t)cB * cT * cD);
  _Float16* KH   = (_Float16*)take(sizeof(_Float16) * (size_t)cB * cT * cD);
  _Float16* VH   = (_Float16*)take(sizeof(_Float16) * (size_t)cB * cT * cD);
  _Float16* OH   = (_Float16*)take(sizeof(_Float16) * (size_t)cB * cT * cD);
  float*    BiasM = (float*)take(sizeof(float) * (size_t)cB * cT * cT);

  auto cvt = [&](const float* s, _Float16* d, int nelem) {
    cvt_f32_f16<<<(nelem + 255) / 256, 256, 0, stream>>>(s, d, nelem);
  };
  cvt(Wq, WqH, cD * 2 * cD);
  cvt(Wk, WkH, cD * 2 * cD);
  cvt(Wv, WvH, cD * cD);
  cvt(Wo, WoH, cD * cD);
  cvt(Wg, WgH, cT * cDM);
  cvt(Gm, GmH, cB * cT * cDM);

  pack_hcat<<<(cB * cT * cD + 255) / 256, 256, 0, stream>>>(Hx, Hf, Hcat);

  const dim3 blk(256);
  // Q, K: (B*T x 2D) @ (D x 2D)^T -> (B*T x D)
  gemm_xwT<true><<<dim3((cB * cT) / 64, cD / 128), blk, 0, stream>>>(
      Hcat, 2 * cD, WqH, bq, QH, cD, 2 * cD);
  gemm_xwT<true><<<dim3((cB * cT) / 64, cD / 128), blk, 0, stream>>>(
      Hcat, 2 * cD, WkH, bk, KH, cD, 2 * cD);
  // V: hx = first D columns of h_cat (lda = 2D, K = D)
  gemm_xwT<true><<<dim3((cB * cT) / 64, cD / 128), blk, 0, stream>>>(
      Hcat, 2 * cD, WvH, bv, VH, cD, cD);
  // bias matrix: (B*T x DM) @ (T x DM)^T + bg -> (B*T x T), reused by all heads
  gemm_xwT<false><<<dim3((cB * cT) / 64, cT / 128), blk, 0, stream>>>(
      GmH, cDM, WgH, bg, BiasM, cT, cDM);

  attn_fwd<<<cB * cH * (cT / 128), blk, 0, stream>>>(QH, KH, VH, BiasM, OH);

  // out projection straight to f32 output
  gemm_xwT<false><<<dim3((cB * cT) / 64, cD / 128), blk, 0, stream>>>(
      OH, cD, WoH, bo, (float*)d_out, cD, cD);
}
